// Attn_5093831213572
// MI455X (gfx1250) — matrix operand order repaired and checked by reading, not compiled, not run on hardware
//
#include <hip/hip_runtime.h>

// ---------------------------------------------------------------------------
// MI455X (gfx1250) fused attention block.
// All GEMMs run on v_wmma_f32_16x16x32_bf16 (f32 accumulate).
// GEMM B-tiles are staged into double-buffered LDS either by the Tensor Data
// Mover (tensor_load_to_lds, TENSORcnt) or by per-lane async copies
// (global_load_async_to_lds_b128, ASYNCcnt), overlapping DMA with WMMA.
// Softmax / RMSNorm statistics stay in f32.
// ---------------------------------------------------------------------------

typedef __attribute__((ext_vector_type(16))) __bf16    v16bf;
typedef __attribute__((ext_vector_type(8)))  float     v8f;
typedef __attribute__((ext_vector_type(4)))  unsigned  v4u;
typedef __attribute__((ext_vector_type(8)))  unsigned  v8u;

#define S_LEN 2048
#define DM    2048
#define NH    16
#define NKV   4
#define HD    128
#define WIN   1024

union BFrag { v16bf v; uint4 u[2]; };

static __device__ inline v8f wmma_bf16(const BFrag& a, const BFrag& b, v8f c) {
  return __builtin_amdgcn_wmma_f32_16x16x32_bf16(
      /*neg_a=*/false, a.v, /*neg_b=*/false, b.v,
      /*c_mod=*/(short)0, c, /*reuse_a=*/false, /*reuse_b=*/false);
}

// --- CDNA5 per-lane async global->LDS copy (16B per lane), ASYNCcnt --------
static __device__ inline void async_ld_b128(const __bf16* gp, __bf16* lp) {
  unsigned laddr = (unsigned)(unsigned long long)lp;  // flat low 32 = LDS off
  asm volatile("global_load_async_to_lds_b128 %0, %1, off"
               :: "v"(laddr), "v"((unsigned long long)gp) : "memory");
}
static __device__ inline void wait_asynccnt0() {
  asm volatile("s_wait_asynccnt 0x0" ::: "memory");
}

// --- CDNA5 Tensor Data Mover: DMA a 128-row x 32-element bf16 tile ---------
// 2D descriptor (groups 0/1 only). LDS padding: tile row = 64B (16 DWORDs,
// pad_interval code 3), pad 16B (4 DWORDs, pad_amount code 3) -> 80B row
// stride in LDS, matching BSTR=40 elements below.
static __device__ inline void tdm_load_tile(const __bf16* gp, unsigned lds_off,
                                            int Kdim) {
  unsigned long long ga = (unsigned long long)gp;
  v4u g0;
  g0.x = 1u;                                             // count = 1 valid D#
  g0.y = lds_off;                                        // lds_addr (bytes)
  g0.z = (unsigned)ga;                                   // global_addr[31:0]
  g0.w = ((unsigned)(ga >> 32) & 0x01FFFFFFu)            // global_addr[56:32]
         | 0x80000000u;                                  // type = 2 ("image")
  v8u g1;
  g1.s0 = (1u << 16)            // data_size = 1 -> 2-byte elements
        | (1u << 20)            // pad_enable
        | (3u << 22)            // pad_interval: 16 DWORDs (64B) between pads
        | (3u << 25);           // pad_amount: 4 DWORDs (16B)
  g1.s1 = ((unsigned)Kdim & 0xFFFFu) << 16;              // tensor_dim0 lo16
  g1.s2 = ((unsigned)Kdim >> 16) | (128u << 16);         // dim0 hi | dim1 lo
  g1.s3 = (32u << 16);                                   // tile_dim0 = 32
  g1.s4 = 128u;                                          // tile_dim1 = 128
  g1.s5 = (unsigned)Kdim;                                // dim0_stride lo32
  g1.s6 = 0u;
  g1.s7 = 0u;
  asm volatile("tensor_load_to_lds %0, %1" :: "s"(g0), "s"(g1) : "memory");
}

// -------------------------- fp32 -> bf16 convert ---------------------------
__global__ __launch_bounds__(256) void cvt_f32_to_bf16(
    const float* __restrict__ in, __bf16* __restrict__ out, int n) {
  int i = (blockIdx.x * 256 + threadIdx.x) * 4;
  if (i < n) {
    float4 f = *reinterpret_cast<const float4*>(in + i);
    out[i + 0] = (__bf16)f.x;
    out[i + 1] = (__bf16)f.y;
    out[i + 2] = (__bf16)f.z;
    out[i + 3] = (__bf16)f.w;
  }
}

// ---------------------------- bf16 WMMA GEMM -------------------------------
// C[M x N] (f32) = A[M x K] (bf16, row-major) @ B[N x K]^T (bf16, row-major).
// Block: 256 threads = 8 waves. Block tile 128(M) x 128(N).
// Wave w owns rows [bm + 16w, +16), full 128-col strip.
// B tile (128 x 32 bf16) staged into double-buffered LDS via TDM or async.
#define BSTR 40                      // padded LDS row stride (elements)
#define TILE_BYTES (128 * BSTR * 2)  // 10240
template <bool USE_TDM>
__global__ __launch_bounds__(256) void gemm_bf16(
    const __bf16* __restrict__ A, const __bf16* __restrict__ B,
    float* __restrict__ C, int M, int N, int K) {
  __shared__ __attribute__((aligned(16))) __bf16 ldsB[2][128 * BSTR];
  const int nTiles = N >> 7;
  const int bm = (blockIdx.x / nTiles) << 7;
  const int bn = (blockIdx.x % nTiles) << 7;
  const int tid  = threadIdx.x;
  const int wv   = tid >> 5;
  const int lane = tid & 31;
  const int hlf  = lane >> 4;
  const int l16  = lane & 15;
  const int mW   = bm + (wv << 4);

  const v8f vzero = {0.f,0.f,0.f,0.f,0.f,0.f,0.f,0.f};
  v8f acc[8];
#pragma unroll
  for (int j = 0; j < 8; ++j) acc[j] = vzero;

  const __bf16* Arow = A + (size_t)(mW + l16) * K;
  const int srow = tid >> 1;            // 128 rows, 2 threads per row
  const int soff = (tid & 1) << 4;      // 16 elements each
  const __bf16* Bst = B + (size_t)(bn + srow) * K + soff;
  const __bf16* Btile = B + (size_t)bn * K;   // TDM tile base (row bn)
  __bf16* lw0 = &ldsB[0][srow * BSTR + soff];
  __bf16* lw1 = &ldsB[1][srow * BSTR + soff];

  // prologue: stage first B tile into buffer 0
  if (USE_TDM) {
    if (wv == 0) tdm_load_tile(Btile, 0u, K);
  } else {
    async_ld_b128(Bst,     lw0);
    async_ld_b128(Bst + 8, lw0 + 8);
  }

  int buf = 0;
  for (int k0 = 0; k0 < K; k0 += 32) {
    if (USE_TDM) {
      if (wv == 0) __builtin_amdgcn_s_wait_tensorcnt(0);  // tile landed
    } else {
      wait_asynccnt0();        // this wave's async writes to ldsB[buf] done
    }
    __syncthreads();           // all writes visible / prev reads retired
    if (k0 + 32 < K) {         // kick off next tile into the other buffer
      if (USE_TDM) {
        if (wv == 0)
          tdm_load_tile(Btile + k0 + 32, buf ? 0u : (unsigned)TILE_BYTES, K);
      } else {
        __bf16* lw = buf ? lw0 : lw1;
        async_ld_b128(Bst + k0 + 32, lw);
        async_ld_b128(Bst + k0 + 40, lw + 8);
      }
    }
    __builtin_prefetch(Arow + k0 + 128, 0, 1);   // global_prefetch_b8
    // A fragment: lane holds row (mW+l16), K runs [hlf*16,+16)
    BFrag af;
    af.u[0] = *reinterpret_cast<const uint4*>(Arow + k0 + hlf * 16);
    af.u[1] = *reinterpret_cast<const uint4*>(Arow + k0 + hlf * 16 + 8);
    // preload all 8 B fragments, then run the WMMA chain
    BFrag bfr[8];
#pragma unroll
    for (int j = 0; j < 8; ++j) {
      const __bf16* bp = &ldsB[buf][(j * 16 + l16) * BSTR + hlf * 16];
      bfr[j].u[0] = *reinterpret_cast<const uint4*>(bp);
      bfr[j].u[1] = *reinterpret_cast<const uint4*>(bp + 8);
    }
#pragma unroll
    for (int j = 0; j < 8; ++j) acc[j] = wmma_bf16(af, bfr[j], acc[j]);
    buf ^= 1;
  }
  // C fragment: element (m = v + 8*hlf, n = l16)
#pragma unroll
  for (int j = 0; j < 8; ++j) {
#pragma unroll
    for (int v = 0; v < 8; ++v) {
      C[(size_t)(mW + v + hlf * 8) * N + bn + j * 16 + l16] = acc[j][v];
    }
  }
}

// ----------------------- RMSNorm + RoPE (q and k) --------------------------
// One wave per (position, head). 128 values -> 4 per lane.
__global__ __launch_bounds__(128) void qk_norm_rope(
    const float* __restrict__ qf, const float* __restrict__ kf,
    const float* __restrict__ qw, const float* __restrict__ kw,
    __bf16* __restrict__ qb, __bf16* __restrict__ kb) {
  const int flat = blockIdx.x * 4 + (threadIdx.x >> 5);
  const int lane = threadIdx.x & 31;
  const int s = flat / (NH + NKV);
  const int r = flat % (NH + NKV);
  const float* src; const float* w; __bf16* dst;
  if (r < NH) {
    src = qf + ((size_t)s * NH + r) * HD;
    w = qw;
    dst = qb + ((size_t)s * NH + r) * HD;
  } else {
    int hh = r - NH;
    src = kf + ((size_t)s * NKV + hh) * HD;
    w = kw;
    dst = kb + ((size_t)s * NKV + hh) * HD;
  }
  const float2* p2 = reinterpret_cast<const float2*>(src);
  float2 a = p2[lane];        // d = 2*lane, 2*lane+1
  float2 b = p2[lane + 32];   // d = 64 + 2*lane, 65 + 2*lane
  float ss = a.x * a.x + a.y * a.y + b.x * b.x + b.y * b.y;
#pragma unroll
  for (int off = 16; off > 0; off >>= 1) ss += __shfl_xor(ss, off, 32);
  float rn = rsqrtf(ss * (1.0f / HD) + 1e-6f);
  const float2* w2 = reinterpret_cast<const float2*>(w);
  float2 wa = w2[lane], wb = w2[lane + 32];
  float x1a = a.x * rn * wa.x;
  float x1b = a.y * rn * wa.y;
  float x2a = b.x * rn * wb.x;
  float x2b = b.y * rn * wb.y;
  // inv_freq[j] = 10000^(-2j/128) = exp(j * -ln(10000)/64)
  const float cth = -0.14391156f;
  float pos = (float)s;
  int j = lane * 2;
  float ang0 = pos * __expf(cth * (float)j);
  float ang1 = pos * __expf(cth * (float)(j + 1));
  float s0, c0, s1, c1;
  __sincosf(ang0, &s0, &c0);
  __sincosf(ang1, &s1, &c1);
  dst[j]      = (__bf16)(x1a * c0 - x2a * s0);
  dst[j + 1]  = (__bf16)(x1b * c1 - x2b * s1);
  dst[j + 64] = (__bf16)(x1a * s0 + x2a * c0);
  dst[j + 65] = (__bf16)(x1b * s1 + x2b * c1);
}

// --------------- V: f32 [s][hkv][d] -> bf16 transposed [hkv][d][s] ---------
__global__ __launch_bounds__(256) void v_to_bf16_t(
    const float* __restrict__ vf, __bf16* __restrict__ vt) {
  int i = blockIdx.x * 256 + threadIdx.x;   // over 2048*512
  int s  = i >> 9;
  int hd = i & 511;
  vt[(size_t)hd * S_LEN + s] = (__bf16)vf[i];
}

// ------------------------- Flash attention ---------------------------------
// One wave per (head, 16-row q tile). Sliding-window causal, online softmax.
#define PSTR 40
__global__ __launch_bounds__(256) void attn_fa(
    const __bf16* __restrict__ qb, const __bf16* __restrict__ kb,
    const __bf16* __restrict__ vt, __bf16* __restrict__ ob) {
  __shared__ __attribute__((aligned(16))) __bf16 ldsP[8 * 16 * PSTR];
  const int tid  = threadIdx.x;
  const int wv   = tid >> 5;
  const int lane = tid & 31;
  const int hlf  = lane >> 4;
  const int l16  = lane & 15;
  const int flat = blockIdx.x * 8 + wv;
  const int h   = flat >> 7;       // 16 heads
  const int qt  = flat & 127;      // 128 q tiles
  const int m0  = qt << 4;
  const int hkv = h >> 2;          // n_rep = 4
  __bf16* lp = &ldsP[wv * 16 * PSTR];

  // Q fragments: 16x128 as 4 WMMA A-fragments (K=32 each)
  BFrag qf4[4];
  const __bf16* Qrow = qb + ((size_t)(m0 + l16) * NH + h) * HD;
#pragma unroll
  for (int c = 0; c < 4; ++c) {
    qf4[c].u[0] = *reinterpret_cast<const uint4*>(Qrow + c * 32 + hlf * 16);
    qf4[c].u[1] = *reinterpret_cast<const uint4*>(Qrow + c * 32 + hlf * 16 + 8);
  }

  const v8f vzero = {0.f,0.f,0.f,0.f,0.f,0.f,0.f,0.f};
  v8f o[8];
#pragma unroll
  for (int j = 0; j < 8; ++j) o[j] = vzero;
  float mrow[8], lrow[8];
#pragma unroll
  for (int v = 0; v < 8; ++v) { mrow[v] = -1e30f; lrow[v] = 0.f; }

  const float scale = 0.08838834764831845f;  // 1/sqrt(128)
  int jb0 = m0 - (WIN - 1);
  if (jb0 < 0) jb0 = 0;
  jb0 &= ~31;
  // jb <= 2016, so key indices jb..jb+31 never exceed 2047 (no OOB).
  for (int jb = jb0; jb < m0 + 16; jb += 32) {
    // ---- scores S = Q K^T for 32 keys (two 16-key subtiles) ----
    v8f sc[2];
#pragma unroll
    for (int t = 0; t < 2; ++t) {
      int key = jb + t * 16 + l16;   // B-frag column = key
      const __bf16* Krow = kb + ((size_t)key * NKV + hkv) * HD;
      BFrag kfr[4];
#pragma unroll
      for (int c = 0; c < 4; ++c) {
        kfr[c].u[0] = *reinterpret_cast<const uint4*>(Krow + c * 32 + hlf * 16);
        kfr[c].u[1] = *reinterpret_cast<const uint4*>(Krow + c * 32 + hlf * 16 + 8);
      }
      v8f s = vzero;
#pragma unroll
      for (int c = 0; c < 4; ++c) s = wmma_bf16(qf4[c], kfr[c], s);
      sc[t] = s;
    }
    // ---- mask, online softmax, write P (bf16) to LDS ----
    float corr[8];
#pragma unroll
    for (int v = 0; v < 8; ++v) {
      int qrow = m0 + v + hlf * 8;
      int key0 = jb + l16;
      int key1 = jb + 16 + l16;
      float s0v = (key0 <= qrow && (qrow - key0) < WIN) ? sc[0][v] * scale : -1e30f;
      float s1v = (key1 <= qrow && (qrow - key1) < WIN) ? sc[1][v] * scale : -1e30f;
      float m2 = fmaxf(s0v, s1v);
#pragma unroll
      for (int off = 8; off > 0; off >>= 1) m2 = fmaxf(m2, __shfl_xor(m2, off, 16));
      float mn = fmaxf(mrow[v], m2);
      float cr = __expf(mrow[v] - mn);
      mrow[v] = mn;
      float p0 = (s0v < -1e29f) ? 0.f : __expf(s0v - mn);
      float p1 = (s1v < -1e29f) ? 0.f : __expf(s1v - mn);
      float ps = p0 + p1;
#pragma unroll
      for (int off = 8; off > 0; off >>= 1) ps += __shfl_xor(ps, off, 16);
      lrow[v] = lrow[v] * cr + ps;
      corr[v] = cr;
      int m = v + hlf * 8;
      lp[m * PSTR + l16]      = (__bf16)p0;
      lp[m * PSTR + 16 + l16] = (__bf16)p1;
    }
#pragma unroll
    for (int j = 0; j < 8; ++j) {
#pragma unroll
      for (int v = 0; v < 8; ++v) o[j][v] *= corr[v];
    }
    __builtin_amdgcn_wave_barrier();
    // ---- P fragment from LDS (A-layout transpose) ----
    BFrag pf;
    pf.u[0] = *reinterpret_cast<const uint4*>(&lp[l16 * PSTR + hlf * 16]);
    pf.u[1] = *reinterpret_cast<const uint4*>(&lp[l16 * PSTR + hlf * 16 + 8]);
    // ---- O += P @ V (V transposed: lane n = head-dim col, K = key run) ----
    int sb = jb + hlf * 16;
    BFrag vfr[8];
#pragma unroll
    for (int j = 0; j < 8; ++j) {
      const __bf16* Vp = vt + (size_t)(hkv * HD + j * 16 + l16) * S_LEN + sb;
      vfr[j].u[0] = *reinterpret_cast<const uint4*>(Vp);
      vfr[j].u[1] = *reinterpret_cast<const uint4*>(Vp + 8);
    }
#pragma unroll
    for (int j = 0; j < 8; ++j) o[j] = wmma_bf16(pf, vfr[j], o[j]);
  }
  // ---- normalize and store bf16 [s][h][d] (== [s][2048] for final GEMM) ----
  float inv[8];
#pragma unroll
  for (int v = 0; v < 8; ++v) inv[v] = 1.0f / lrow[v];
#pragma unroll
  for (int j = 0; j < 8; ++j) {
#pragma unroll
    for (int v = 0; v < 8; ++v) {
      int m = m0 + v + hlf * 8;
      ob[((size_t)m * NH + h) * HD + j * 16 + l16] = (__bf16)(o[j][v] * inv[v]);
    }
  }
}

// ---------------------------------------------------------------------------
extern "C" void kernel_launch(void* const* d_in, const int* in_sizes, int n_in,
                              void* d_out, int out_size, void* d_ws, size_t ws_size,
                              hipStream_t stream) {
  const float* x  = (const float*)d_in[0];
  const float* wq = (const float*)d_in[1];
  const float* wk = (const float*)d_in[2];
  const float* wv = (const float*)d_in[3];
  const float* wo = (const float*)d_in[4];
  const float* qw = (const float*)d_in[5];
  const float* kw = (const float*)d_in[6];
  float* out = (float*)d_out;

  char* ws = (char*)d_ws;
  size_t off = 0;
  auto carve = [&](size_t bytes) {
    char* p = ws + off;
    off += (bytes + 255) & ~(size_t)255;
    return p;
  };
  __bf16* xb    = (__bf16*)carve((size_t)S_LEN * DM * 2);      // x bf16
  __bf16* wqb   = (__bf16*)carve((size_t)DM * DM * 2);         // wq bf16
  __bf16* wkb   = (__bf16*)carve((size_t)NKV * HD * DM * 2);   // wk bf16
  __bf16* wvb   = (__bf16*)carve((size_t)NKV * HD * DM * 2);   // wv bf16
  __bf16* wob   = (__bf16*)carve((size_t)DM * DM * 2);         // wo bf16
  float*  qf    = (float*) carve((size_t)S_LEN * DM * 4);      // q f32
  float*  kf    = (float*) carve((size_t)S_LEN * NKV * HD * 4);
  float*  vf    = (float*) carve((size_t)S_LEN * NKV * HD * 4);
  __bf16* qbn   = (__bf16*)carve((size_t)S_LEN * DM * 2);      // q norm+rope bf16
  __bf16* kbn   = (__bf16*)carve((size_t)S_LEN * NKV * HD * 2);
  __bf16* vt    = (__bf16*)carve((size_t)S_LEN * NKV * HD * 2);// V^T bf16
  __bf16* attnb = (__bf16*)carve((size_t)S_LEN * DM * 2);      // attn out bf16
  (void)ws_size; (void)in_sizes; (void)n_in; (void)out_size;

  // 1) fp32 -> bf16 conversions
  cvt_f32_to_bf16<<<(S_LEN * DM) / 1024, 256, 0, stream>>>(x,  xb,  S_LEN * DM);
  cvt_f32_to_bf16<<<(DM * DM)    / 1024, 256, 0, stream>>>(wq, wqb, DM * DM);
  cvt_f32_to_bf16<<<(NKV*HD*DM)  / 1024, 256, 0, stream>>>(wk, wkb, NKV * HD * DM);
  cvt_f32_to_bf16<<<(NKV*HD*DM)  / 1024, 256, 0, stream>>>(wv, wvb, NKV * HD * DM);
  cvt_f32_to_bf16<<<(DM * DM)    / 1024, 256, 0, stream>>>(wo, wob, DM * DM);

  // 2) QKV projections (WMMA GEMM, TDM-staged B tiles)
  gemm_bf16<true><<<(S_LEN/128)*(DM/128),       256, 0, stream>>>(xb, wqb, qf, S_LEN, DM,       DM);
  gemm_bf16<true><<<(S_LEN/128)*((NKV*HD)/128), 256, 0, stream>>>(xb, wkb, kf, S_LEN, NKV * HD, DM);
  gemm_bf16<true><<<(S_LEN/128)*((NKV*HD)/128), 256, 0, stream>>>(xb, wvb, vf, S_LEN, NKV * HD, DM);

  // 3) RMSNorm + RoPE on q,k ; V transpose+convert
  qk_norm_rope<<<(S_LEN * (NH + NKV)) / 4, 128, 0, stream>>>(qf, kf, qw, kw, qbn, kbn);
  v_to_bf16_t<<<(S_LEN * NKV * HD) / 256, 256, 0, stream>>>(vf, vt);

  // 4) sliding-window flash attention (WMMA)
  attn_fa<<<(NH * (S_LEN / 16)) / 8, 256, 0, stream>>>(qbn, kbn, vt, attnb);

  // 5) output projection -> f32 result (async-LDS staged GEMM path)
  gemm_bf16<false><<<(S_LEN/128)*(DM/128), 256, 0, stream>>>(attnb, wob, out, S_LEN, DM, DM);
}
